// TSP_Decoder_28647431864773
// MI455X (gfx1250) — matrix operand order repaired and checked by reading, not compiled, not run on hardware
//
#include <hip/hip_runtime.h>

typedef __attribute__((ext_vector_type(16))) __bf16 v16bf;
typedef __attribute__((ext_vector_type(8)))  __bf16 v8bf;
typedef __attribute__((ext_vector_type(8)))  float  v8f;
typedef __attribute__((ext_vector_type(4)))  unsigned int u32x4;
typedef __attribute__((ext_vector_type(8)))  int i32x8;
typedef __attribute__((ext_vector_type(4)))  int i32x4;

#define WMMA_BF16(a, b, c) \
  __builtin_amdgcn_wmma_f32_16x16x32_bf16(false, (a), false, (b), (short)0, (c), false, false)

// Problem sizes (fixed by the reference)
#define BB 64
#define PP 100
#define NN 1000
#define EE 512
#define HH 8
#define KDD 64
#define MM 10
#define HK 512          // H*KD
#define BP 6400         // B*P

// ---------------------------------------------------------------------------
// TDM helpers: 2D fp32 tile load, global -> LDS (descriptor per ISA ch.8)
// ---------------------------------------------------------------------------
static __device__ __forceinline__ unsigned lds_addr_u32(void* p) {
  __attribute__((address_space(3))) void* lp =
      (__attribute__((address_space(3))) void*)p;
  return (unsigned)(unsigned long long)lp;
}

// Load a tile_d1 x tile_d0 (rows x cols) fp32 tile whose top-left element is at
// gaddr, from a row-major tensor with tensor_d0 valid cols, tensor_d1 valid
// rows and row stride stride_d0 (elements). OOB reads are zero-filled by TDM.
static __device__ __forceinline__ void tdm_load_2d_f32(
    unsigned lds_byte_addr, const void* gaddr,
    unsigned tensor_d0, unsigned tensor_d1, unsigned stride_d0,
    unsigned tile_d0, unsigned tile_d1)
{
  const unsigned long long ga = (unsigned long long)gaddr;
  u32x4 g0;
  g0[0] = 1u;                                        // count=1 (user mode)
  g0[1] = lds_byte_addr;                             // lds_addr
  g0[2] = (unsigned)(ga & 0xFFFFFFFFu);              // global_addr[31:0]
  g0[3] = ((unsigned)(ga >> 32) & 0x01FFFFFFu) | (2u << 30);  // addr[56:32], type=2
  i32x8 g1;
  g1[0] = (int)(2u << 16);                           // data_size=4B, mask=0
  g1[1] = (int)((tensor_d0 & 0xFFFFu) << 16);        // tensor_dim0[15:0]
  g1[2] = (int)(((tensor_d0 >> 16) & 0xFFFFu) | ((tensor_d1 & 0xFFFFu) << 16));
  g1[3] = (int)(((tensor_d1 >> 16) & 0xFFFFu) | ((tile_d0 & 0xFFFFu) << 16));
  g1[4] = (int)(tile_d1 & 0xFFFFu);                  // tile_dim1 (tile_dim2=0)
  g1[5] = (int)stride_d0;                            // tensor_dim0_stride lo32
  g1[6] = 0;                                         // stride hi / dim1_stride
  g1[7] = 0;
  i32x4 g2 = {0, 0, 0, 0};
  i32x4 g3 = {0, 0, 0, 0};
#if __clang_major__ >= 23
  i32x8 g4 = {0, 0, 0, 0, 0, 0, 0, 0};
  __builtin_amdgcn_tensor_load_to_lds(g0, g1, g2, g3, g4, 0);
#else
  __builtin_amdgcn_tensor_load_to_lds(g0, g1, g2, g3, 0);
#endif
}

// ---------------------------------------------------------------------------
// Kernel 1: q = heads(q_last_in @ Wq^T) + q_info + q_first  -> bf16 [B,H,P,KD]
// ---------------------------------------------------------------------------
__global__ __launch_bounds__(256) void k1_build_q(
    const float* __restrict__ enc_node, const float* __restrict__ enc_ag,
    const float* __restrict__ lengths, const float* __restrict__ max_dis,
    const float* __restrict__ remain_max_dis, const float* __restrict__ q_first,
    const float* __restrict__ Wq, const float* __restrict__ dis_emb,
    const int* __restrict__ depot_num, const int* __restrict__ route_cnt,
    const int* __restrict__ left_city, const int* __restrict__ city_num_p,
    __bf16* __restrict__ q_out)
{
  const int lane = threadIdx.x & 31;
  const int wv   = threadIdx.x >> 5;
  const int tile = blockIdx.x * 8 + wv;      // 400*32 = 12800 tiles
  const int tm   = tile >> 5;                // M tile (rows of 16)
  const int tn   = tile & 31;                // N tile (cols of 16)
  const int l16  = lane & 15;
  const int col  = tn * 16 + l16;
  const int kA0  = (lane < 16) ? 0 : 16;
  const int kB0  = (lane < 16) ? 0 : 16;
  const int rowA = tm * 16 + l16;
  const int half8 = (lane < 16) ? 0 : 8;

  v8f acc = {};
  for (int kc = 0; kc < 32; ++kc) {
    const float* src = (kc < 16) ? enc_node : enc_ag;
    const float* ap  = src + (size_t)rowA * EE + (kc & 15) * 32 + kA0;
    v16bf a;
#pragma unroll
    for (int j = 0; j < 8; ++j) a[j]     = (__bf16)ap[j];
#pragma unroll
    for (int j = 0; j < 8; ++j) a[8 + j] = (__bf16)ap[8 + j];
    const float* bp = Wq + (size_t)col * 1026 + kc * 32 + kB0;
    v16bf bb;
#pragma unroll
    for (int j = 0; j < 16; ++j) bb[j] = (__bf16)bp[j];
    acc = WMMA_BF16(a, bb, acc);
  }

  const float w_frac = Wq[(size_t)col * 1026 + 1024];
  const float w_lc   = Wq[(size_t)col * 1026 + 1025];
  const float d0 = dis_emb[col * 4 + 0];
  const float d1 = dis_emb[col * 4 + 1];
  const float d2 = dis_emb[col * 4 + 2];
  const float d3 = dis_emb[col * 4 + 3];
  const int   h  = col >> 6, kd = col & 63;
  const float inv_city = 1.0f / (float)city_num_p[0];
  const int   rbase = tm * 16 + half8;
#pragma unroll
  for (int i = 0; i < 8; ++i) {
    const int r  = rbase + i;
    const int b  = r / PP, p = r - b * PP;
    const int rc = route_cnt[r];
    const int rn = depot_num[b] - 1;
    float s = acc[i];
    s += (1.0f - (float)(rc + 1) / (float)(rn + 1)) * w_frac;
    s += ((float)left_city[r] * inv_city) * w_lc;
    s += lengths[r * MM + rc] * d0 + lengths[r * MM + rn] * d1
       + max_dis[r] * d2 + remain_max_dis[r] * d3;
    const size_t qi = (((size_t)b * HH + h) * PP + p) * KDD + kd;
    s += q_first[qi];
    q_out[qi] = (__bf16)s;
  }
}

// ---------------------------------------------------------------------------
// Kernel 2: flash attention per (b,h). 7 waves, each owns a 16-row M tile.
// K/V 32x64 fp32 tiles are DMA'd into double-buffered LDS by the Tensor Data
// Mover (wave 0 issues, s_wait_tensorcnt + barrier publishes); next chunk's
// DMA overlaps with current chunk's WMMAs. OOB rows are zero-filled by TDM.
// ---------------------------------------------------------------------------
__global__ __launch_bounds__(224) void k2_attn(
    const __bf16* __restrict__ qbf,   // [B,H,P,KD]
    const float*  __restrict__ kg,    // [B,H,N,KD]
    const float*  __restrict__ vg,    // [B,H,N,KD]
    const float*  __restrict__ ninf,  // [B,P,N]
    __bf16* __restrict__ attn)        // [B,P,H*KD]
{
  __shared__ __align__(16) float  k_buf[2][32][64];   // 16 KB
  __shared__ __align__(16) float  v_buf[2][32][64];   // 16 KB
  __shared__ __align__(16) __bf16 p_lds[7][16][32];   // 7 KB

  const int bh   = blockIdx.x;
  const int b    = bh >> 3, h = bh & 7;
  const int tid  = threadIdx.x;
  const int lane = tid & 31, wv = tid >> 5;
  const int m0   = wv * 16;
  const int l16  = lane & 15;
  const int kA0  = (lane < 16) ? 0 : 16;
  const int kB0  = (lane < 16) ? 0 : 16;
  const int half8 = (lane < 16) ? 0 : 8;

  // Preload Q A-operands (2 K-chunks of 32 over KD=64)
  v16bf aq[2];
  {
    const int p = m0 + l16;
    const __bf16* qp = qbf + (((size_t)b * HH + h) * PP + p) * KDD;
#pragma unroll
    for (int kk = 0; kk < 2; ++kk) {
      v16bf a = {};
      if (p < PP) {
        const v8bf lo = *(const v8bf*)(qp + kk * 32 + kA0);
        const v8bf hi = *(const v8bf*)(qp + kk * 32 + kA0 + 8);
#pragma unroll
        for (int j = 0; j < 8; ++j) { a[j] = lo[j]; a[8 + j] = hi[j]; }
      }
      aq[kk] = a;
    }
  }

  v8f acc[4] = {v8f{}, v8f{}, v8f{}, v8f{}};
  float mrow[8], lrow[8];
#pragma unroll
  for (int i = 0; i < 8; ++i) { mrow[i] = -3.0e38f; lrow[i] = 0.0f; }

  const size_t kvbase = ((size_t)b * HH + h) * NN * KDD;
  const float* kgb = kg + kvbase;
  const float* vgb = vg + kvbase;

  // Prime the pipeline: DMA chunk 0 into buffer 0
  if (wv == 0) {
    tdm_load_2d_f32(lds_addr_u32(&k_buf[0][0][0]), kgb, 64, NN, 64, 64, 32);
    tdm_load_2d_f32(lds_addr_u32(&v_buf[0][0][0]), vgb, 64, NN, 64, 64, 32);
  }

  for (int nc = 0; nc < 32; ++nc) {
    const int cur = nc & 1;
    const int n0  = nc * 32;
    if (wv == 0) __builtin_amdgcn_s_wait_tensorcnt(0);
    __syncthreads();   // publish chunk nc's LDS; previous compute done
    if (wv == 0 && nc + 1 < 32) {
      const int n1 = n0 + 32;
      tdm_load_2d_f32(lds_addr_u32(&k_buf[cur ^ 1][0][0]),
                      kgb + (size_t)n1 * KDD, 64, NN - n1, 64, 64, 32);
      tdm_load_2d_f32(lds_addr_u32(&v_buf[cur ^ 1][0][0]),
                      vgb + (size_t)n1 * KDD, 64, NN - n1, 64, 64, 32);
    }

    // Scores for two 16-col subtiles (B-operand: fp32 LDS -> bf16 in regs)
    float sv[2][8];
#pragma unroll
    for (int sub = 0; sub < 2; ++sub) {
      const float* kp = &k_buf[cur][sub * 16 + l16][0];
      v16bf b0, b1;
#pragma unroll
      for (int j = 0; j < 16; ++j) b0[j] = (__bf16)kp[kB0 + j];        // kd 0..31
#pragma unroll
      for (int j = 0; j < 16; ++j) b1[j] = (__bf16)kp[32 + kB0 + j];   // kd 32..63
      v8f s = {};
      s = WMMA_BF16(aq[0], b0, s);
      s = WMMA_BF16(aq[1], b1, s);
      const int scol = n0 + sub * 16 + l16;
#pragma unroll
      for (int i = 0; i < 8; ++i) {
        float x = s[i] * 0.125f;               // 1/sqrt(KD)
        const int pr = m0 + half8 + i;
        if (scol < NN) {
          if (pr < PP) x += ninf[((size_t)(b * PP + pr)) * NN + scol];
        } else {
          x = -1.0e30f;
        }
        sv[sub][i] = x;
      }
    }

    // Row max (reduce across the 16 lanes holding each row)
    float corr[8];
#pragma unroll
    for (int i = 0; i < 8; ++i) {
      float mx = fmaxf(sv[0][i], sv[1][i]);
#pragma unroll
      for (int msk = 1; msk < 16; msk <<= 1) mx = fmaxf(mx, __shfl_xor(mx, msk, 32));
      const float mn = fmaxf(mrow[i], mx);
      corr[i] = __expf(mrow[i] - mn);
      mrow[i] = mn;
    }
    // p = exp(s - m), stash to LDS in C-layout as bf16
#pragma unroll
    for (int sub = 0; sub < 2; ++sub) {
#pragma unroll
      for (int i = 0; i < 8; ++i) {
        const float pv = __expf(sv[sub][i] - mrow[i]);
        sv[sub][i] = pv;
        p_lds[wv][half8 + i][sub * 16 + l16] = (__bf16)pv;
      }
    }
#pragma unroll
    for (int i = 0; i < 8; ++i) {
      float s2 = sv[0][i] + sv[1][i];
#pragma unroll
      for (int msk = 1; msk < 16; msk <<= 1) s2 += __shfl_xor(s2, msk, 32);
      lrow[i] = lrow[i] * corr[i] + s2;
#pragma unroll
      for (int t = 0; t < 4; ++t) acc[t][i] *= corr[i];
    }

    // P back out of LDS in A-layout, then PV: 4 d-tiles of 16
    v16bf pa;
    {
      const __bf16* pp = &p_lds[wv][l16][0];
      const v8bf lo = *(const v8bf*)(pp + kA0);
      const v8bf hi = *(const v8bf*)(pp + kA0 + 8);
#pragma unroll
      for (int j = 0; j < 8; ++j) { pa[j] = lo[j]; pa[8 + j] = hi[j]; }
    }
#pragma unroll
    for (int t = 0; t < 4; ++t) {
      v16bf vb;
#pragma unroll
      for (int j = 0; j < 16; ++j)
        vb[j] = (__bf16)v_buf[cur][kB0 + j][t * 16 + l16];
      acc[t] = WMMA_BF16(pa, vb, acc[t]);
    }
  }

  // Normalize and store bf16 [B,P,H*KD]
#pragma unroll
  for (int t = 0; t < 4; ++t) {
#pragma unroll
    for (int i = 0; i < 8; ++i) {
      const int pr = m0 + half8 + i;
      if (pr < PP) {
        const float o = acc[t][i] / lrow[i];
        attn[((size_t)(b * PP + pr)) * HK + h * KDD + t * 16 + l16] = (__bf16)o;
      }
    }
  }
}

// ---------------------------------------------------------------------------
// Kernel 3a: mh = attn @ mh_w^T + mh_b   ([6400,512] x [512,512] -> bf16)
// ---------------------------------------------------------------------------
__global__ __launch_bounds__(256) void k3a_mh(
    const __bf16* __restrict__ attn, const float* __restrict__ mh_w,
    const float* __restrict__ mh_b, __bf16* __restrict__ mh_out)
{
  const int lane = threadIdx.x & 31;
  const int wv   = threadIdx.x >> 5;
  const int tile = blockIdx.x * 8 + wv;     // 400*32 tiles
  const int tm   = tile >> 5;
  const int tn   = tile & 31;
  const int l16  = lane & 15;
  const int col  = tn * 16 + l16;
  const int kA0  = (lane < 16) ? 0 : 16;
  const int kB0  = (lane < 16) ? 0 : 16;
  const int rowA = tm * 16 + l16;
  const int half8 = (lane < 16) ? 0 : 8;

  v8f acc = {};
  for (int kc = 0; kc < 16; ++kc) {
    const __bf16* ap = attn + (size_t)rowA * HK + kc * 32;
    const v8bf lo = *(const v8bf*)(ap + kA0);
    const v8bf hi = *(const v8bf*)(ap + kA0 + 8);
    v16bf a;
#pragma unroll
    for (int j = 0; j < 8; ++j) { a[j] = lo[j]; a[8 + j] = hi[j]; }
    const float* bp = mh_w + (size_t)col * EE + kc * 32 + kB0;
    v16bf bb;
#pragma unroll
    for (int j = 0; j < 16; ++j) bb[j] = (__bf16)bp[j];
    acc = WMMA_BF16(a, bb, acc);
  }
  const float bias = mh_b[col];
#pragma unroll
  for (int i = 0; i < 8; ++i) {
    const int r = tm * 16 + half8 + i;
    mh_out[(size_t)r * EE + col] = (__bf16)(acc[i] + bias);
  }
}

// ---------------------------------------------------------------------------
// Kernel 3b: score2[b,p,n] = mh[b,p,:] . shk[b,:,n]. Per-batch GEMM; the
// 32(e) x 128(n) fp32 shk tile is TDM-DMA'd into double-buffered LDS, with
// the next K-chunk's DMA overlapping the current chunk's WMMAs.
// grid = B * 7 ptiles * 8 ngroups.
// ---------------------------------------------------------------------------
__global__ __launch_bounds__(256) void k3b_score2(
    const __bf16* __restrict__ mh,   // [6400,512]
    const float*  __restrict__ shk,  // [B,512,1000]
    float* __restrict__ score2)      // [6400,1000]
{
  __shared__ __align__(16) float b_lds[2][32][128];   // 32 KB

  const int blk = blockIdx.x;        // b*56 + pt*8 + ng
  const int ng  = blk & 7;
  const int pt  = (blk >> 3) % 7;
  const int b   = blk / 56;
  const int tid = threadIdx.x, lane = tid & 31, wv = tid >> 5;
  const int l16 = lane & 15;
  const int kA0 = (lane < 16) ? 0 : 16;
  const int kB0 = (lane < 16) ? 0 : 16;
  const int half8 = (lane < 16) ? 0 : 8;
  const int ntile = ng * 8 + wv;     // 0..63, valid < 63
  const int prow  = pt * 16 + l16;
  const float* shkb = shk + (size_t)b * EE * NN;
  const unsigned ncols = (unsigned)(NN - ng * 128);   // valid cols in tile

  if (wv == 0) {
    tdm_load_2d_f32(lds_addr_u32(&b_lds[0][0][0]),
                    shkb + ng * 128, ncols, EE, NN, 128, 32);
  }

  v8f acc = {};
  for (int kc = 0; kc < 16; ++kc) {
    const int cur = kc & 1;
    if (wv == 0) __builtin_amdgcn_s_wait_tensorcnt(0);
    __syncthreads();
    if (wv == 0 && kc + 1 < 16) {
      tdm_load_2d_f32(lds_addr_u32(&b_lds[cur ^ 1][0][0]),
                      shkb + (size_t)(kc + 1) * 32 * NN + ng * 128,
                      ncols, EE - (kc + 1) * 32, NN, 128, 32);
    }
    if (ntile < 63) {
      v16bf a = {};
      if (prow < PP) {
        const __bf16* ap = mh + ((size_t)(b * PP + prow)) * EE + kc * 32;
        const v8bf lo = *(const v8bf*)(ap + kA0);
        const v8bf hi = *(const v8bf*)(ap + kA0 + 8);
#pragma unroll
        for (int j = 0; j < 8; ++j) { a[j] = lo[j]; a[8 + j] = hi[j]; }
      }
      v16bf bb;
#pragma unroll
      for (int j = 0; j < 16; ++j)
        bb[j] = (__bf16)b_lds[cur][kB0 + j][wv * 16 + l16];
      acc = WMMA_BF16(a, bb, acc);
    }
  }
  if (ntile < 63) {
    const int ncol = ntile * 16 + l16;
#pragma unroll
    for (int i = 0; i < 8; ++i) {
      const int pr = pt * 16 + half8 + i;
      if (pr < PP && ncol < NN)
        score2[((size_t)(b * PP + pr)) * NN + ncol] = acc[i];
    }
  }
}

// ---------------------------------------------------------------------------
// Kernel 4: final distance-biased tanh clip + softmax. One block per (b,p).
// ---------------------------------------------------------------------------
__device__ __forceinline__ float warp_max(float v) {
#pragma unroll
  for (int m = 1; m < 32; m <<= 1) v = fmaxf(v, __shfl_xor(v, m, 32));
  return v;
}
__device__ __forceinline__ float warp_sum(float v) {
#pragma unroll
  for (int m = 1; m < 32; m <<= 1) v += __shfl_xor(v, m, 32);
  return v;
}

__global__ __launch_bounds__(256) void k4_final(
    const float* __restrict__ score2, const float* __restrict__ cur_dist,
    const float* __restrict__ ninf, const float* __restrict__ dist_alpha,
    float* __restrict__ out)
{
  __shared__ float red[8];
  const int r   = blockIdx.x;
  const int tid = threadIdx.x;
  const float* cd = cur_dist + (size_t)r * NN;
  const float* sc = score2 + (size_t)r * NN;
  const float* mk = ninf + (size_t)r * NN;
  float* op = out + (size_t)r * NN;
  const float alpha = dist_alpha[0];
  const float inv_sqrt_e = 0.04419417382415922f;   // 1/sqrt(512)

  // 1) max(cur_dist) over the row
  float mx = -3.0e38f;
  for (int n = tid; n < NN; n += 256) mx = fmaxf(mx, cd[n]);
  mx = warp_max(mx);
  if ((tid & 31) == 0) red[tid >> 5] = mx;
  __syncthreads();
  float mdist = red[0];
#pragma unroll
  for (int i = 1; i < 8; ++i) mdist = fmaxf(mdist, red[i]);
  __syncthreads();
  const float inv_md = 1.0f / mdist;

  // 2) clipped score + mask, track max
  float vals[4];
  float vmax = -3.0e38f;
  for (int n = tid, j = 0; n < NN; n += 256, ++j) {
    const float v = 10.0f * tanhf(sc[n] * inv_sqrt_e - alpha * cd[n] * inv_md) + mk[n];
    vals[j] = v;
    vmax = fmaxf(vmax, v);
  }
  vmax = warp_max(vmax);
  if ((tid & 31) == 0) red[tid >> 5] = vmax;
  __syncthreads();
  float gmax = red[0];
#pragma unroll
  for (int i = 1; i < 8; ++i) gmax = fmaxf(gmax, red[i]);
  __syncthreads();

  // 3) exp + sum
  float s = 0.0f;
  for (int n = tid, j = 0; n < NN; n += 256, ++j) {
    const float e = __expf(vals[j] - gmax);
    vals[j] = e;
    s += e;
  }
  s = warp_sum(s);
  if ((tid & 31) == 0) red[tid >> 5] = s;
  __syncthreads();
  float gsum = red[0];
#pragma unroll
  for (int i = 1; i < 8; ++i) gsum += red[i];
  const float inv_s = 1.0f / gsum;

  // 4) normalize
  for (int n = tid, j = 0; n < NN; n += 256, ++j) op[n] = vals[j] * inv_s;
}

// ---------------------------------------------------------------------------
extern "C" void kernel_launch(void* const* d_in, const int* in_sizes, int n_in,
                              void* d_out, int out_size, void* d_ws, size_t ws_size,
                              hipStream_t stream) {
  (void)in_sizes; (void)n_in; (void)out_size; (void)ws_size;
  const float* enc_node   = (const float*)d_in[0];
  const float* enc_ag     = (const float*)d_in[1];
  const float* cur_dist   = (const float*)d_in[2];
  const float* ninf_mask  = (const float*)d_in[3];
  const float* lengths    = (const float*)d_in[4];
  const float* max_dis    = (const float*)d_in[5];
  const float* remain_md  = (const float*)d_in[6];
  const float* q_first    = (const float*)d_in[7];
  const float* kten       = (const float*)d_in[8];
  const float* vten       = (const float*)d_in[9];
  const float* shk        = (const float*)d_in[10];
  const float* Wq         = (const float*)d_in[11];
  const float* dis_emb    = (const float*)d_in[12];
  const float* mh_w       = (const float*)d_in[13];
  const float* mh_b       = (const float*)d_in[14];
  const float* dist_alpha = (const float*)d_in[15];
  const int*   depot_num  = (const int*)d_in[16];
  const int*   route_cnt  = (const int*)d_in[17];
  const int*   left_city  = (const int*)d_in[18];
  const int*   city_num   = (const int*)d_in[19];
  float* out = (float*)d_out;

  // Workspace layout (bytes)
  char* ws = (char*)d_ws;
  __bf16* q_bf    = (__bf16*)(ws + 0);          // 6,553,600
  __bf16* attn_bf = (__bf16*)(ws + 6553600);    // 6,553,600
  __bf16* mh_bf   = (__bf16*)(ws + 13107200);   // 6,553,600
  float*  score2  = (float*)(ws + 19660800);    // 25,600,000

  k1_build_q<<<1600, 256, 0, stream>>>(enc_node, enc_ag, lengths, max_dis,
      remain_md, q_first, Wq, dis_emb, depot_num, route_cnt, left_city,
      city_num, q_bf);
  k2_attn<<<BB * HH, 224, 0, stream>>>(q_bf, kten, vten, ninf_mask, attn_bf);
  k3a_mh<<<1600, 256, 0, stream>>>(attn_bf, mh_w, mh_b, mh_bf);
  k3b_score2<<<BB * 7 * 8, 256, 0, stream>>>(mh_bf, shk, score2);
  k4_final<<<BP, 256, 0, stream>>>(score2, cur_dist, ninf_mask, dist_alpha, out);
}
